// XL_encoder_71390946394722
// MI455X (gfx1250) — compile-verified, matrix operand order checked by reading, not run on hardware
//
#include <hip/hip_runtime.h>
#include <hip/hip_bf16.h>
#include <math.h>

// ---------------------------------------------------------------------------
// Types & helpers
// ---------------------------------------------------------------------------
typedef __attribute__((ext_vector_type(16))) __bf16 v16bf;
typedef __attribute__((ext_vector_type(8)))  __bf16 v8bf;
typedef __attribute__((ext_vector_type(8)))  float  v8f;

#define DEV __device__ __forceinline__

DEV unsigned short f2bf_bits(float x) {
  union { float f; unsigned u; } c; c.f = x;
  unsigned u = c.u;
  u += 0x7FFFu + ((u >> 16) & 1u);       // round-to-nearest-even
  return (unsigned short)(u >> 16);
}
// Fragment gather per CDNA5 16-bit A/B VGPR layout:
// lanes 0-15 take K = {0..7, 16..23}, lanes 16-31 take K = {8..15, 24..31}.
DEV v16bf fragld(const unsigned short* rowbase, int lko) {
  const __bf16* p = reinterpret_cast<const __bf16*>(rowbase);
  v8bf lo = *reinterpret_cast<const v8bf*>(p + lko);
  v8bf hi = *reinterpret_cast<const v8bf*>(p + 16 + lko);
  return __builtin_shufflevector(lo, hi, 0,1,2,3,4,5,6,7,8,9,10,11,12,13,14,15);
}
// Async DMA global->LDS, 16B per lane (ASYNCcnt-tracked). IOFFSET is added to
// BOTH the LDS and global byte address (ISA 15.18.3), so base + offset:16
// copies 32 contiguous bytes.
DEV void async_b128(unsigned ldsOff, const unsigned short* g) {
  asm volatile("global_load_async_to_lds_b128 %0, %1, off"
               :: "v"(ldsOff), "v"(g) : "memory");
}
DEV void async_b128_o16(unsigned ldsOff, const unsigned short* g) {
  asm volatile("global_load_async_to_lds_b128 %0, %1, off offset:16"
               :: "v"(ldsOff), "v"(g) : "memory");
}
DEV void wait_async0() { asm volatile("s_wait_asynccnt 0" ::: "memory"); }
DEV unsigned ldsAddr(const void* p) { return (unsigned)(uintptr_t)p; }

DEV float sigmoidf_(float x) { return 1.f / (1.f + __expf(-x)); }
DEV float siluf_(float x)    { return x * sigmoidf_(x); }
DEV float logsigf_(float x)  { return fminf(x, 0.f) - log1pf(__expf(-fabsf(x))); }

DEV float blockReduceSum(float v, float* sbuf) {
  int lane = threadIdx.x & 31, w = threadIdx.x >> 5;
  #pragma unroll
  for (int o = 16; o > 0; o >>= 1) v += __shfl_xor(v, o, 32);
  if (lane == 0) sbuf[w] = v;
  __syncthreads();
  int nw = (blockDim.x + 31) >> 5;
  v = (threadIdx.x < nw) ? sbuf[threadIdx.x] : 0.f;
  if (w == 0) {
    #pragma unroll
    for (int o = 16; o > 0; o >>= 1) v += __shfl_xor(v, o, 32);
    if (lane == 0) sbuf[0] = v;
  }
  __syncthreads();
  float r = sbuf[0];
  __syncthreads();
  return r;
}

// ---------------------------------------------------------------------------
// f32 -> bf16 streaming converters (run once per tensor, outside hot loops)
// ---------------------------------------------------------------------------
__global__ void cvt_k(const float* __restrict__ x, unsigned short* __restrict__ y,
                      long n, float s) {
  long i4 = ((long)blockIdx.x * blockDim.x + threadIdx.x) * 4;
  if (i4 >= n) return;
  float a0 = x[i4] * s, a1 = x[i4 + 1] * s, a2 = x[i4 + 2] * s, a3 = x[i4 + 3] * s;
  uint2 r;
  r.x = (unsigned)f2bf_bits(a0) | ((unsigned)f2bf_bits(a1) << 16);
  r.y = (unsigned)f2bf_bits(a2) | ((unsigned)f2bf_bits(a3) << 16);
  *reinterpret_cast<uint2*>(y + i4) = r;
}
// (K,N) f32 -> (N,K) bf16 transpose-convert (weights)
__global__ void cvtT_k(const float* __restrict__ w, unsigned short* __restrict__ y,
                       int K, int N, long total) {
  long idx = (long)blockIdx.x * blockDim.x + threadIdx.x;
  if (idx >= total) return;
  int k = (int)(idx % K); long n = idx / K;
  y[idx] = f2bf_bits(w[(size_t)k * N + n]);
}

// ---------------------------------------------------------------------------
// WMMA GEMM: C[M,N] = act(A[M,K] @ W^T + bias) (+ res).
// A: (M,K) bf16 row-major. W: (N,K) bf16 row-major. Tiles staged into LDS via
// async DMA (global_load_async_to_lds_b128). M%128==0, N%128==0, K%32==0.
// ---------------------------------------------------------------------------
__global__ __launch_bounds__(256)
void gemm_bf16(const unsigned short* __restrict__ A, const unsigned short* __restrict__ W,
               const float* __restrict__ bias, const float* __restrict__ res,
               float* __restrict__ C, int M, int N, int K, int act) {
  __shared__ unsigned short lA[128][40];   // 128 x 32 bf16, 80B padded rows
  __shared__ unsigned short lB[128][40];
  (void)M;
  int tid = threadIdx.x;
  int m0 = blockIdx.y * 128, n0 = blockIdx.x * 128;
  int wave = tid >> 5, lane = tid & 31;
  int wm = (wave >> 2) * 64, wn = (wave & 3) * 32;
  int lrow = lane & 15, lko = (lane & 16) ? 8 : 0, rsel = (lane >> 4) * 8;

  // staging map: thread -> (row, 32B half of the 64B row)
  int srow = tid >> 1, shalf = tid & 1;
  unsigned lofsA = ldsAddr(&lA[0][0]) + srow * 80 + shalf * 32;
  unsigned lofsB = ldsAddr(&lB[0][0]) + srow * 80 + shalf * 32;
  const unsigned short* aRow = A + (size_t)(m0 + srow) * K + shalf * 16;
  const unsigned short* wRow = W + (size_t)(n0 + srow) * K + shalf * 16;

  v8f acc[4][2] = {};
  for (int k0 = 0; k0 < K; k0 += 32) {
    __syncthreads();
    async_b128(lofsA, aRow + k0);
    async_b128_o16(lofsA, aRow + k0);
    async_b128(lofsB, wRow + k0);
    async_b128_o16(lofsB, wRow + k0);
    wait_async0();
    __syncthreads();
    v16bf af[4], bfr[2];
    #pragma unroll
    for (int mi = 0; mi < 4; mi++) af[mi] = fragld(&lA[wm + mi * 16 + lrow][0], lko);
    #pragma unroll
    for (int ni = 0; ni < 2; ni++) bfr[ni] = fragld(&lB[wn + ni * 16 + lrow][0], lko);
    #pragma unroll
    for (int mi = 0; mi < 4; mi++)
      #pragma unroll
      for (int ni = 0; ni < 2; ni++)
        acc[mi][ni] = __builtin_amdgcn_wmma_f32_16x16x32_bf16(
            false, af[mi], false, bfr[ni], (short)0, acc[mi][ni], false, false);
  }
  #pragma unroll
  for (int mi = 0; mi < 4; mi++) {
    #pragma unroll
    for (int ni = 0; ni < 2; ni++) {
      int gn = n0 + wn + ni * 16 + lrow;
      float bv = bias ? bias[gn] : 0.f;
      int gmB = m0 + wm + mi * 16 + rsel;
      #pragma unroll
      for (int v = 0; v < 8; v++) {
        float x = acc[mi][ni][v] + bv;
        if (act == 1) x = fmaxf(x, 0.f);
        size_t off = (size_t)(gmB + v) * N + gn;
        if (res) x += res[off];
        C[off] = x;
      }
    }
  }
}

// ---------------------------------------------------------------------------
// mLSTM parallel attention (flash-style, running max of the gate matrix logD).
// q,k,v: (B,S,NH*DH) bf16 (q pre-scaled by dh^-0.5); ipre: (B,NH,S) f32;
// lfc: (B,NH,S+1) f32; out: (B,S,NH*DH) f32.
// ---------------------------------------------------------------------------
template<int DH>
__global__ __launch_bounds__(256)
void mlstm_attn(const unsigned short* __restrict__ q, const unsigned short* __restrict__ k,
                const unsigned short* __restrict__ v, const float* __restrict__ ipre,
                const float* __restrict__ lfc, float* __restrict__ out,
                int S, int NH) {
  __shared__ unsigned short lK[32][DH + 8];   // keys row-major
  __shared__ unsigned short lVT[DH][40];      // V transposed (dh x 32 keys)
  __shared__ unsigned short lP[8][16][40];    // per-wave P staging (16 x 32)
  __shared__ float lLf[32], lIp[32];

  const int inner = NH * DH;
  int bh = blockIdx.y; int b = bh / NH, h = bh % NH;
  int qblk = blockIdx.x * 128;
  const unsigned short* qp = q + (size_t)b * S * inner + h * DH;
  const unsigned short* kp = k + (size_t)b * S * inner + h * DH;
  const unsigned short* vp = v + (size_t)b * S * inner + h * DH;
  const float* ip = ipre + ((size_t)b * NH + h) * S;
  const float* lf = lfc + ((size_t)b * NH + h) * (S + 1);

  int tid = threadIdx.x, wave = tid >> 5, lane = tid & 31;
  int lrow = lane & 15, lko = (lane & 16) ? 8 : 0, rsel = (lane >> 4) * 8;
  int qrow0 = qblk + wave * 16;

  // Q fragments straight from global bf16 (kept in registers all K-blocks)
  v16bf qf[DH / 32];
  {
    const unsigned short* pq = qp + (size_t)(qrow0 + lrow) * inner;
    #pragma unroll
    for (int c = 0; c < DH / 32; c++) qf[c] = fragld(pq + c * 32, lko);
  }
  float lfq[8], mrow[8], lrw[8];
  #pragma unroll
  for (int vv = 0; vv < 8; vv++) {
    lfq[vv] = lf[qrow0 + vv + rsel + 1];
    mrow[vv] = -1e30f; lrw[vv] = 0.f;
  }
  v8f accN[DH / 16] = {};

  const unsigned ldsK0 = ldsAddr(&lK[0][0]);
  const int rowChunks = DH / 8;               // 16B chunks per key row
  int nkb = (qblk + 128) / 32;
  for (int kb = 0; kb < nkb; kb++) {
    int kt0 = kb * 32;
    __syncthreads();
    // K tile via async DMA (row-major copy)
    for (int ch = tid; ch < 32 * rowChunks; ch += 256) {
      int r = ch / rowChunks, cq = ch % rowChunks;
      async_b128(ldsK0 + (unsigned)(r * (DH + 8) * 2 + cq * 16),
                 kp + (size_t)(kt0 + r) * inner + cq * 8);
    }
    // V tile transposed (bf16 element copy)
    for (int idx = tid; idx < 32 * DH; idx += 256) {
      int r = idx / DH, c2 = idx % DH;
      lVT[c2][r] = vp[(size_t)(kt0 + r) * inner + c2];
    }
    if (tid < 32) { lLf[tid] = lf[kt0 + tid + 1]; lIp[tid] = ip[kt0 + tid]; }
    wait_async0();
    __syncthreads();
    if (kt0 > qrow0 + 15) continue;   // wave-uniform: EXEC stays all-ones

    // ---- scores: two 16x16 tiles over 32 keys ----
    v8f sc0 = {}, sc1 = {};
    #pragma unroll
    for (int c = 0; c < DH / 32; c++) {
      v16bf k0f = fragld(&lK[lrow][c * 32], lko);
      v16bf k1f = fragld(&lK[16 + lrow][c * 32], lko);
      sc0 = __builtin_amdgcn_wmma_f32_16x16x32_bf16(false, qf[c], false, k0f, (short)0, sc0, false, false);
      sc1 = __builtin_amdgcn_wmma_f32_16x16x32_bf16(false, qf[c], false, k1f, (short)0, sc1, false, false);
    }
    float lfk0 = lLf[lrow],      ipk0 = lIp[lrow];
    float lfk1 = lLf[16 + lrow], ipk1 = lIp[16 + lrow];
    int kt_0 = kt0 + lrow, kt_1 = kt0 + 16 + lrow;
    float scl[8];
    #pragma unroll
    for (int vv = 0; vv < 8; vv++) {
      int qs = qrow0 + vv + rsel;
      float ld0 = (kt_0 <= qs) ? (lfq[vv] - lfk0 + ipk0) : -1e30f;
      float ld1 = (kt_1 <= qs) ? (lfq[vv] - lfk1 + ipk1) : -1e30f;
      float rm = fmaxf(ld0, ld1);
      #pragma unroll
      for (int o = 1; o < 16; o <<= 1) rm = fmaxf(rm, __shfl_xor(rm, o, 32));
      float mn = fmaxf(mrow[vv], rm);
      float sf = __expf(mrow[vv] - mn);
      mrow[vv] = mn; scl[vv] = sf;
      float p0 = sc0[vv] * __expf(ld0 - mn);
      float p1 = sc1[vv] * __expf(ld1 - mn);
      float rs = p0 + p1;                     // signed row-sum (|sum C| norm)
      #pragma unroll
      for (int o = 1; o < 16; o <<= 1) rs += __shfl_xor(rs, o, 32);
      lrw[vv] = lrw[vv] * sf + rs;
      lP[wave][vv + rsel][lrow]      = f2bf_bits(p0);
      lP[wave][vv + rsel][16 + lrow] = f2bf_bits(p1);
    }
    #pragma unroll
    for (int n = 0; n < DH / 16; n++)
      #pragma unroll
      for (int vv = 0; vv < 8; vv++) accN[n][vv] *= scl[vv];
    asm volatile("s_wait_dscnt 0" ::: "memory");   // P staging RAW fence
    v16bf pa = fragld(&lP[wave][lrow][0], lko);
    #pragma unroll
    for (int n = 0; n < DH / 16; n++) {
      v16bf vf = fragld(&lVT[n * 16 + lrow][0], lko);
      accN[n] = __builtin_amdgcn_wmma_f32_16x16x32_bf16(
          false, pa, false, vf, (short)0, accN[n], false, false);
    }
  }
  float nrm[8];
  #pragma unroll
  for (int vv = 0; vv < 8; vv++)
    nrm[vv] = fmaxf(fabsf(lrw[vv]), __expf(-mrow[vv])) + 1e-6f;
  #pragma unroll
  for (int n = 0; n < DH / 16; n++) {
    #pragma unroll
    for (int vv = 0; vv < 8; vv++) {
      int row = qrow0 + vv + rsel;
      out[((size_t)b * S + row) * inner + h * DH + n * 16 + lrow] = accN[n][vv] / nrm[vv];
    }
  }
}

// ---------------------------------------------------------------------------
// sLSTM recurrent scan. Block per (b,h), 4*DH threads. Sequential over S.
// ---------------------------------------------------------------------------
template<int DH>
__global__ __launch_bounds__(4 * DH)
void slstm_scan(const float* __restrict__ xi, const float* __restrict__ xf,
                const float* __restrict__ xz, const float* __restrict__ xo,
                const float* __restrict__ R, const float* __restrict__ bias,
                float* __restrict__ hs, int S, int NH) {
  __shared__ float sh[DH];
  __shared__ float sr[4 * DH];
  int bh = blockIdx.x; int b = bh / NH, h = bh % NH;
  int t = threadIdx.x;
  const int F = NH * DH;
  const float* Rh = R + (size_t)h * DH * 4 * DH;
  float bi = 0, bfv = 0, bz = 0, bo = 0, c = 0, nn = 0, mm = 0;
  if (t < DH) {
    bi  = bias[(0 * NH + h) * DH + t];
    bfv = bias[(1 * NH + h) * DH + t];
    bz  = bias[(2 * NH + h) * DH + t];
    bo  = bias[(3 * NH + h) * DH + t];
    sh[t] = 0.f;
  }
  __syncthreads();
  for (int s = 0; s < S; s++) {
    float r = 0.f;
    #pragma unroll 4
    for (int d = 0; d < DH; d++) r += sh[d] * Rh[(size_t)d * (4 * DH) + t];
    sr[t] = r;
    __syncthreads();
    if (t < DH) {
      size_t xoff = ((size_t)b * S + s) * F + h * DH + t;
      float it  = xi[xoff] + sr[t] + bi;
      float lgf = mm + logsigf_(xf[xoff] + sr[DH + t] + bfv);
      float mn  = fmaxf(it, lgf);
      float ia  = __expf(it - mn);
      float fa  = __expf(lgf - mn);
      c  = fa * c + ia * tanhf(xz[xoff] + sr[2 * DH + t] + bz);
      nn = fa * nn + ia;
      mm = mn;
      float hv = sigmoidf_(xo[xoff] + sr[3 * DH + t] + bo) * c / nn;
      sh[t] = hv;
      hs[xoff] = hv;
    }
    __syncthreads();
  }
}

// ---------------------------------------------------------------------------
// Elementwise / reduction kernels
// ---------------------------------------------------------------------------
__global__ void layernorm_k(const float* __restrict__ x, const float* __restrict__ w,
                            float* __restrict__ y, int D) {
  __shared__ float sbuf[8];
  size_t row = blockIdx.x;
  const float* xr = x + row * D;
  float s = 0;
  for (int i = threadIdx.x; i < D; i += blockDim.x) s += xr[i];
  float mean = blockReduceSum(s, sbuf) / D;
  float vs = 0;
  for (int i = threadIdx.x; i < D; i += blockDim.x) { float d = xr[i] - mean; vs += d * d; }
  float var = blockReduceSum(vs, sbuf) / D;
  float inv = rsqrtf(var + 1e-5f);
  for (int i = threadIdx.x; i < D; i += blockDim.x) y[row * D + i] = (xr[i] - mean) * inv * w[i];
}

__global__ void convsilu_k(const float* __restrict__ x, int ldx,
                           const float* __restrict__ w, const float* __restrict__ bb,
                           float* __restrict__ y, int S, int F, long total) {
  long idx = (long)blockIdx.x * blockDim.x + threadIdx.x;
  if (idx >= total) return;
  int f = (int)(idx % F); long bs = idx / F; int s = (int)(bs % S); long bn = bs / S;
  float a = bb[f];
  #pragma unroll
  for (int j = 0; j < 4; j++) {
    int sp = s - 3 + j;
    if (sp >= 0) a += x[((size_t)bn * S + sp) * ldx + f] * w[f * 4 + j];
  }
  y[((size_t)bn * S + s) * F + f] = siluf_(a);
}

__global__ void headwise_k(const float* __restrict__ x, int ldx,
                           const float* __restrict__ w, float* __restrict__ y,
                           int NB, int BS, long rows) {
  long idx = (long)blockIdx.x * blockDim.x + threadIdx.x;
  int F = NB * BS;
  if (idx >= rows * F) return;
  int o = (int)(idx % BS); int nb = (int)((idx / BS) % NB); long row = idx / F;
  const float* xr = x + row * (size_t)ldx + nb * BS;
  const float* wr = w + ((size_t)nb * BS + o) * BS;
  float a = 0;
  for (int i = 0; i < BS; i++) a += xr[i] * wr[i];
  y[row * (size_t)F + nb * BS + o] = a;
}

__global__ void mgates_k(const float* __restrict__ q, const float* __restrict__ k,
                         const float* __restrict__ v, int inner,
                         const float* __restrict__ igw, const float* __restrict__ igb,
                         const float* __restrict__ fgw, const float* __restrict__ fgb,
                         float* __restrict__ ipre, float* __restrict__ fpre,
                         int S, int NH, long rows) {
  long idx = (long)blockIdx.x * blockDim.x + threadIdx.x;
  if (idx >= rows * NH) return;
  int hh = (int)(idx % NH); long row = idx / NH;
  long b = row / S; int s = (int)(row % S);
  float ai = igb[hh], af = fgb[hh];
  for (int j = 0; j < inner; j++) {
    float qv = q[row * inner + j], kv = k[row * inner + j], vv = v[row * inner + j];
    ai += qv * igw[(size_t)j * NH + hh] + kv * igw[(size_t)(inner + j) * NH + hh]
        + vv * igw[(size_t)(2 * inner + j) * NH + hh];
    af += qv * fgw[(size_t)j * NH + hh] + kv * fgw[(size_t)(inner + j) * NH + hh]
        + vv * fgw[(size_t)(2 * inner + j) * NH + hh];
  }
  ipre[((size_t)b * NH + hh) * S + s] = ai;
  fpre[((size_t)b * NH + hh) * S + s] = af;
}

__global__ void lfc_k(const float* __restrict__ fpre, float* __restrict__ lfc, int S, int BH) {
  int i = blockIdx.x * blockDim.x + threadIdx.x;
  if (i >= BH) return;
  const float* f = fpre + (size_t)i * S;
  float* o = lfc + (size_t)i * (S + 1);
  float acc = 0; o[0] = 0;
  for (int s = 0; s < S; s++) { acc += logsigf_(f[s]); o[s + 1] = acc; }
}

__global__ void mpost_k(const float* __restrict__ ht, const float* __restrict__ onw,
                        const float* __restrict__ skip, const float* __restrict__ xconv,
                        const float* __restrict__ z, int ldz,
                        float* __restrict__ out, int NH, int DH) {
  __shared__ float sbuf[8];
  int blk = blockIdx.x; int hh = blk % NH; size_t row = blk / NH;
  int F = NH * DH; int i = threadIdx.x;
  float x = ht[row * F + hh * DH + i];
  float mean = blockReduceSum(x, sbuf) / DH;
  float d = x - mean;
  float var = blockReduceSum(d * d, sbuf) / DH;
  float g = d * rsqrtf(var + 1e-5f) * onw[hh * DH + i];
  int col = hh * DH + i;
  float hv = g + skip[col] * xconv[row * F + col];
  float zv = z[row * (size_t)ldz + col];
  out[row * F + col] = hv * siluf_(zv);
}

__global__ void spost_k(const float* __restrict__ hsv, const float* __restrict__ gnw,
                        const float* __restrict__ xres, float* __restrict__ out,
                        int NH, int DH) {
  __shared__ float sbuf[8];
  int blk = blockIdx.x; int hh = blk % NH; size_t row = blk / NH;
  int F = NH * DH; int i = threadIdx.x;
  float x = hsv[row * F + hh * DH + i];
  float mean = blockReduceSum(x, sbuf) / DH;
  float d = x - mean;
  float var = blockReduceSum(d * d, sbuf) / DH;
  float g = d * rsqrtf(var + 1e-5f) * gnw[hh * DH + i];
  int col = hh * DH + i;
  out[row * F + col] = xres[row * F + col] + g;
}

__global__ void geluup_k(const float* __restrict__ u, int ldu, int up,
                         float* __restrict__ act, long rows) {
  long idx = (long)blockIdx.x * blockDim.x + threadIdx.x;
  if (idx >= rows * up) return;
  int j = (int)(idx % up); long row = idx / up;
  float g = u[row * (size_t)ldu + j], uv = u[row * (size_t)ldu + up + j];
  act[idx] = 0.5f * g * (1.f + erff(g * 0.70710678f)) * uv;
}

__global__ void conv1_k(const float* __restrict__ x, const float* __restrict__ w,
                        const float* __restrict__ bb, float* __restrict__ y,
                        int N, long total) {
  long idx = (long)blockIdx.x * blockDim.x + threadIdx.x;
  if (idx >= total) return;
  int c = (int)(idx % 128); long bn = idx / 128; int n = (int)(bn % N); long b = bn / N;
  float a = bb[c];
  #pragma unroll
  for (int j = 0; j < 3; j++) a += x[((size_t)b * 3 + j) * N + n] * w[c * 3 + j];
  y[idx] = fmaxf(a, 0.f);
}

__global__ void maxpool_k(const float* __restrict__ x, float* __restrict__ y,
                          int S, int C, int B) {
  int i = blockIdx.x * blockDim.x + threadIdx.x;
  if (i >= B * C) return;
  int c = i % C; int b = i / C;
  float m = -3.4e38f;
  for (int s = 0; s < S; s++) m = fmaxf(m, x[((size_t)b * S + s) * C + c]);
  y[i] = m;
}

__global__ void concat_k(const float* __restrict__ h2, const float* __restrict__ gf,
                         float* __restrict__ y, int S, int C, long total) {
  long idx = (long)blockIdx.x * blockDim.x + threadIdx.x;
  if (idx >= total) return;
  int c = (int)(idx % (2 * C)); long bs = idx / (2 * C); int b = (int)(bs / S);
  y[idx] = (c < C) ? h2[bs * (size_t)C + c] : gf[(size_t)b * C + (c - C)];
}

// ---------------------------------------------------------------------------
// Host orchestration
// ---------------------------------------------------------------------------
struct MP { const float *ln,*proj_up,*conv_w,*conv_b,*q_w,*k_w,*v_w,*ig_w,*ig_b,
                        *fg_w,*fg_b,*outnorm,*skip,*proj_down; };
struct SP { const float *ln1,*conv_w,*conv_b,*wi,*wf,*wz,*wo,*R,*bias,*gn,*ln2,
                        *ff_up,*ff_down; };
struct STK { MP m[3]; SP s; const float* post_ln; };

extern "C" void kernel_launch(void* const* d_in, const int* in_sizes, int n_in,
                              void* d_out, int out_size, void* d_ws, size_t ws_size,
                              hipStream_t stream) {
  (void)in_sizes; (void)n_in; (void)out_size; (void)ws_size;
  const int Bb = 2, S = 2048, RS = Bb * S;      // RS = 4096 rows
  int ii = 0;
  const float* X       = (const float*)d_in[ii++];
  const float* conv1_w = (const float*)d_in[ii++];
  const float* conv1_b = (const float*)d_in[ii++];
  const float* conv2_w = (const float*)d_in[ii++];
  const float* conv2_b = (const float*)d_in[ii++];
  const float* conv3_w = (const float*)d_in[ii++];
  const float* conv3_b = (const float*)d_in[ii++];
  const float* conv4_w = (const float*)d_in[ii++];
  const float* conv4_b = (const float*)d_in[ii++];
  STK st[2];
  for (int si = 0; si < 2; si++) {
    for (int mi = 0; mi < 3; mi++) {
      MP& m = st[si].m[mi];
      m.ln = (const float*)d_in[ii++];      m.proj_up = (const float*)d_in[ii++];
      m.conv_w = (const float*)d_in[ii++];  m.conv_b = (const float*)d_in[ii++];
      m.q_w = (const float*)d_in[ii++];     m.k_w = (const float*)d_in[ii++];
      m.v_w = (const float*)d_in[ii++];     m.ig_w = (const float*)d_in[ii++];
      m.ig_b = (const float*)d_in[ii++];    m.fg_w = (const float*)d_in[ii++];
      m.fg_b = (const float*)d_in[ii++];    m.outnorm = (const float*)d_in[ii++];
      m.skip = (const float*)d_in[ii++];    m.proj_down = (const float*)d_in[ii++];
    }
    SP& s = st[si].s;
    s.ln1 = (const float*)d_in[ii++];   s.conv_w = (const float*)d_in[ii++];
    s.conv_b = (const float*)d_in[ii++];s.wi = (const float*)d_in[ii++];
    s.wf = (const float*)d_in[ii++];    s.wz = (const float*)d_in[ii++];
    s.wo = (const float*)d_in[ii++];    s.R = (const float*)d_in[ii++];
    s.bias = (const float*)d_in[ii++];  s.gn = (const float*)d_in[ii++];
    s.ln2 = (const float*)d_in[ii++];   s.ff_up = (const float*)d_in[ii++];
    s.ff_down = (const float*)d_in[ii++];
    st[si].post_ln = (const float*)d_in[ii++];
  }

  // Workspace arena: f32 region then bf16 (ushort) region
  float* W0 = (float*)d_ws;
  size_t off = 0;
  auto alloc = [&](size_t n) { float* p = W0 + off; off += n; return p; };
  float* xbuf  = alloc((size_t)RS * 512);
  float* xalt  = alloc((size_t)RS * 512);
  float* lnb   = alloc((size_t)RS * 512);
  float* up    = alloc((size_t)RS * 2048);
  float* xconv = alloc((size_t)RS * 1024);
  float* qb    = alloc((size_t)RS * 1024);
  float* kb    = alloc((size_t)RS * 1024);
  float* vb    = alloc((size_t)RS * 1024);
  float* htb   = alloc((size_t)RS * 1024);
  float* hgb   = alloc((size_t)RS * 1024);
  float* ipb   = alloc((size_t)Bb * 4 * S);
  float* fpb   = alloc((size_t)Bb * 4 * S);
  float* lfb   = alloc((size_t)Bb * 4 * (S + 1));
  float* gfb   = alloc(512);
  unsigned short* U0 = (unsigned short*)(W0 + off);
  size_t uoff = 0;
  auto ualloc = [&](size_t n) { unsigned short* p = U0 + uoff; uoff += n; return p; };
  unsigned short* abf = ualloc((size_t)RS * 1024);   // GEMM A operand (bf16)
  unsigned short* wbf = ualloc((size_t)2048 * 512);  // GEMM W operand (N,K bf16)
  unsigned short* qbf = ualloc((size_t)RS * 1024);
  unsigned short* kbf = ualloc((size_t)RS * 1024);
  unsigned short* vbf = ualloc((size_t)RS * 1024);

  auto grid1 = [](long tot) { return dim3((unsigned)((tot + 255) / 256)); };
  // GEMM with on-the-fly operand conversion. wNK: W already (N,K) row-major.
  auto gemm = [&](const float* A, const float* Wf, bool wNK, const float* bias,
                  const float* res, float* C, int M, int N, int K, int act) {
    cvt_k<<<grid1((long)M * K / 4), 256, 0, stream>>>(A, abf, (long)M * K, 1.f);
    if (wNK) cvt_k<<<grid1((long)N * K / 4), 256, 0, stream>>>(Wf, wbf, (long)N * K, 1.f);
    else     cvtT_k<<<grid1((long)N * K), 256, 0, stream>>>(Wf, wbf, K, N, (long)N * K);
    dim3 g(N / 128, M / 128);
    gemm_bf16<<<g, 256, 0, stream>>>(abf, wbf, bias, res, C, M, N, K, act);
  };

  auto mlstm = [&](float* x, int D, const MP& p) {
    int inner = 2 * D, NH = 4, DH = inner / NH, ld2 = 2 * inner;
    layernorm_k<<<RS, 256, 0, stream>>>(x, p.ln, lnb, D);
    gemm(lnb, p.proj_up, false, nullptr, nullptr, up, RS, 2 * inner, D, 0);
    long ti = (long)RS * inner;
    convsilu_k<<<grid1(ti), 256, 0, stream>>>(up, ld2, p.conv_w, p.conv_b, xconv, S, inner, ti);
    headwise_k<<<grid1(ti), 256, 0, stream>>>(xconv, inner, p.q_w, qb, inner / 4, 4, RS);
    headwise_k<<<grid1(ti), 256, 0, stream>>>(xconv, inner, p.k_w, kb, inner / 4, 4, RS);
    headwise_k<<<grid1(ti), 256, 0, stream>>>(up, ld2, p.v_w, vb, inner / 4, 4, RS);
    long tg = (long)RS * NH;
    mgates_k<<<grid1(tg), 256, 0, stream>>>(qb, kb, vb, inner, p.ig_w, p.ig_b,
                                            p.fg_w, p.fg_b, ipb, fpb, S, NH, RS);
    lfc_k<<<1, 32, 0, stream>>>(fpb, lfb, S, Bb * NH);
    float qscale = 1.f / sqrtf((float)DH);
    cvt_k<<<grid1(ti / 4), 256, 0, stream>>>(qb, qbf, ti, qscale);
    cvt_k<<<grid1(ti / 4), 256, 0, stream>>>(kb, kbf, ti, 1.f);
    cvt_k<<<grid1(ti / 4), 256, 0, stream>>>(vb, vbf, ti, 1.f);
    dim3 ag(S / 128, Bb * NH);
    if (DH == 64) mlstm_attn<64><<<ag, 256, 0, stream>>>(qbf, kbf, vbf, ipb, lfb, htb, S, NH);
    else          mlstm_attn<256><<<ag, 256, 0, stream>>>(qbf, kbf, vbf, ipb, lfb, htb, S, NH);
    mpost_k<<<RS * NH, DH, 0, stream>>>(htb, p.outnorm, p.skip, xconv, up + inner, ld2,
                                        hgb, NH, DH);
    gemm(hgb, p.proj_down, false, nullptr, x, x, RS, D, inner, 0);
  };

  auto slstm = [&](float* x, int D, const SP& p) {
    int NH = 4, DH = D / 4;
    layernorm_k<<<RS, 256, 0, stream>>>(x, p.ln1, lnb, D);
    long td = (long)RS * D;
    convsilu_k<<<grid1(td), 256, 0, stream>>>(lnb, D, p.conv_w, p.conv_b, xconv, S, D, td);
    headwise_k<<<grid1(td), 256, 0, stream>>>(xconv, D, p.wi, qb, NH, DH, RS);
    headwise_k<<<grid1(td), 256, 0, stream>>>(xconv, D, p.wf, kb, NH, DH, RS);
    headwise_k<<<grid1(td), 256, 0, stream>>>(lnb, D, p.wz, vb, NH, DH, RS);
    headwise_k<<<grid1(td), 256, 0, stream>>>(lnb, D, p.wo, htb, NH, DH, RS);
    if (DH == 32) slstm_scan<32><<<Bb * NH, 128, 0, stream>>>(qb, kb, vb, htb, p.R, p.bias, hgb, S, NH);
    else          slstm_scan<128><<<Bb * NH, 512, 0, stream>>>(qb, kb, vb, htb, p.R, p.bias, hgb, S, NH);
    spost_k<<<RS * NH, DH, 0, stream>>>(hgb, p.gn, x, x, NH, DH);
  };

  auto ffn = [&](float* x, int D, const SP& p) {
    int updim = 64 * ((D * 13 + 639) / 640);   // 64*ceil(1.3*D/64): 192 / 704
    layernorm_k<<<RS, 256, 0, stream>>>(x, p.ln2, lnb, D);
    gemm(lnb, p.ff_up, false, nullptr, nullptr, up, RS, 2 * updim, D, 0);
    long tu = (long)RS * updim;
    geluup_k<<<grid1(tu), 256, 0, stream>>>(up, 2 * updim, updim, xconv, RS);
    gemm(xconv, p.ff_down, false, nullptr, x, x, RS, D, updim, 0);
  };

  auto stack = [&](float* x, int D, const STK& sp) {
    mlstm(x, D, sp.m[0]);
    slstm(x, D, sp.s);
    ffn(x, D, sp.s);
    mlstm(x, D, sp.m[1]);
    mlstm(x, D, sp.m[2]);
    layernorm_k<<<RS, 256, 0, stream>>>(x, sp.post_ln, lnb, D);  // result -> lnb
  };

  // ---- forward ----
  const int N = 2048;
  long t1 = (long)Bb * N * 128;
  conv1_k<<<grid1(t1), 256, 0, stream>>>(X, conv1_w, conv1_b, xbuf, N, t1);
  stack(xbuf, 128, st[0]);                                        // -> lnb (B,S,128)
  gemm(lnb, conv2_w, true, conv2_b, nullptr, up, RS, 256, 128, 0);
  maxpool_k<<<grid1(Bb * 256), 256, 0, stream>>>(up, gfb, S, 256, Bb);
  long tc = (long)RS * 512;
  concat_k<<<grid1(tc), 256, 0, stream>>>(up, gfb, xbuf, S, 256, tc);
  gemm(xbuf, conv3_w, true, conv3_b, nullptr, xalt, RS, 512, 512, 1);
  stack(xalt, 512, st[1]);                                        // -> lnb (B,S,512)
  gemm(lnb, conv4_w, true, conv4_b, nullptr, htb, RS, 1024, 512, 0);
  maxpool_k<<<grid1(Bb * 1024), 256, 0, stream>>>(htb, (float*)d_out, S, 1024, Bb);
}
